// SparseMSARowAttentionWithPairBias_75282186764799
// MI455X (gfx1250) — matrix operand order repaired and checked by reading, not compiled, not run on hardware
//
#include <hip/hip_runtime.h>
#include <hip/hip_bf16.h>

// ---------------------------------------------------------------------------
// Problem constants: B=1, S=32, N=512, C_M=256, C_Z=128, H=8, D=32,
// half-window 16, G=16 global tokens.
// ---------------------------------------------------------------------------
#define SS 32
#define NN 512
#define CM 256
#define CZ 128
#define HH 8
#define DD 32
#define NEGV (-1e9f)

typedef __attribute__((ext_vector_type(16))) __bf16        v16bf;
typedef __attribute__((ext_vector_type(8)))  float         v8f;
typedef __attribute__((ext_vector_type(4)))  unsigned int  u32x4;
typedef __attribute__((ext_vector_type(4)))  float         f32x4;
typedef __attribute__((ext_vector_type(2)))  float         f32x2;

union AF { v16bf v; unsigned short u[16]; u32x4 q[2]; };
union CF { v8f   v; float          e[8];  };

// float -> bf16 bits, round-to-nearest-even
__device__ __forceinline__ unsigned short f2bf(float f) {
    unsigned int x = __float_as_uint(f);
    unsigned int r = (x + 0x7fffu + ((x >> 16) & 1u)) >> 16;
    return (unsigned short)r;
}

// 16-bit A-matrix (16x32) fragment: lane group g holds two contiguous K runs:
// K = 8g + [0..7]  (elements 0..7)  and  K = 16 + 8g + [0..7] (elements 8..15)
// B-matrix (32x16): lane holds col N = lane%16, K order the same as A's
// -> both operands load as 2 x b128 when laid out along K.

// ---------------------------------------------------------------------------
// Kernel 1: bf16 conversions + K-major (transposed) weight layouts.
//   wqkvT: [col=768][k=256]  (cols 0..255=q, 256..511=k, 512..767=v)
//   woT:   [col=256][k=256]
//   wbT:   [h=16][c=128]     (heads 8..15 zero)
// ---------------------------------------------------------------------------
__global__ void prep_kernel(const float* __restrict__ msa,
                            const float* __restrict__ w_q,
                            const float* __restrict__ w_k,
                            const float* __restrict__ w_v,
                            const float* __restrict__ w_o,
                            const float* __restrict__ w_b,
                            unsigned short* msa_bf, unsigned short* wqkvT,
                            unsigned short* woT, unsigned short* wbT) {
    const int stride = gridDim.x * blockDim.x;
    const int tid = blockIdx.x * blockDim.x + threadIdx.x;
    // packed 2-wide conversion of msa
    for (int i = tid; i < (SS * NN * CM) / 2; i += stride) {
        f32x2 f = ((const f32x2*)msa)[i];
        unsigned int pk = (unsigned int)f2bf(f.x) | ((unsigned int)f2bf(f.y) << 16);
        ((unsigned int*)msa_bf)[i] = pk;
    }
    for (int i = tid; i < 3 * CM * CM; i += stride) {
        int col = i >> 8, k = i & 255;
        int which = col >> 8, cc = col & 255;
        const float* w = (which == 0) ? w_q : ((which == 1) ? w_k : w_v);
        wqkvT[i] = f2bf(w[k * CM + cc]);
    }
    for (int i = tid; i < CM * CM; i += stride) {
        int col = i >> 8, k = i & 255;
        woT[i] = f2bf(w_o[k * CM + col]);
    }
    for (int i = tid; i < 16 * CZ; i += stride) {
        int h = i >> 7, c = i & 127;
        wbT[i] = f2bf(h < HH ? w_b[c * HH + h] : 0.f);
    }
}

// ---------------------------------------------------------------------------
// Kernel 2: fused QKV projection WMMA GEMM, fully unrolled K=256 chain.
//   q scaled by 1/sqrt(D). q,k stored [s][h][n][d]; v stored TRANSPOSED
//   [s][h][d][n] so attention PV B-fragments are contiguous along keys.
// ---------------------------------------------------------------------------
__global__ void qkv_gemm_kernel(const unsigned short* __restrict__ msa_bf,
                                const unsigned short* __restrict__ wqkvT,
                                const float* __restrict__ b_q,
                                const float* __restrict__ b_k,
                                const float* __restrict__ b_v,
                                unsigned short* q_ws, unsigned short* k_ws,
                                unsigned short* vT_ws) {
    const int L = threadIdx.x & 31;
    const int wave = threadIdx.x >> 5;
    const int rt = blockIdx.x;                 // row tile 0..1023
    const int ct = blockIdx.y * 8 + wave;      // col tile 0..47
    const int g = L >> 4, m = L & 15;

    CF c;
#pragma unroll
    for (int r = 0; r < 8; ++r) c.e[r] = 0.f;

    const int row = rt * 16 + m;
    const int col = ct * 16 + m;
    const unsigned short* arow = msa_bf + row * CM;
    const unsigned short* brow = wqkvT + col * CM;
#pragma unroll
    for (int kb = 0; kb < CM; kb += 32) {
        AF a, b;
        a.q[0] = *(const u32x4*)(arow + kb + 8 * g);
        a.q[1] = *(const u32x4*)(arow + kb + 16 + 8 * g);
        b.q[0] = *(const u32x4*)(brow + kb + 8 * g);
        b.q[1] = *(const u32x4*)(brow + kb + 16 + 8 * g);
        c.v = __builtin_amdgcn_wmma_f32_16x16x32_bf16(false, a.v, false, b.v,
                                                      (short)0, c.v, false, false);
    }

    const int which = col >> 8, cc = col & 255;
    const float* bias = (which == 0) ? b_q : ((which == 1) ? b_k : b_v);
    const float scal = (which == 0) ? 0.1767766953f : 1.0f;  // 1/sqrt(32)
    const int h = cc >> 5, d = cc & 31;
#pragma unroll
    for (int r = 0; r < 8; ++r) {
        int grow = rt * 16 + r + 8 * g;
        int s = grow >> 9, n = grow & 511;
        unsigned short val = f2bf((c.e[r] + bias[cc]) * scal);
        if (which == 0)      q_ws [(((s * HH + h) * NN) + n) * DD + d] = val;
        else if (which == 1) k_ws [(((s * HH + h) * NN) + n) * DD + d] = val;
        else                 vT_ws[(((s * HH + h) * DD) + d) * NN + n] = val;
    }
}

// ---------------------------------------------------------------------------
// Kernel 3: pair-bias WMMA at mask-live tile pairs only (~15% of pair).
// ---------------------------------------------------------------------------
__global__ void bias_kernel(const float* __restrict__ pair,
                            const unsigned short* __restrict__ wbT,
                            const float* __restrict__ b_b,
                            float* bias_ws) {
    const int t = blockIdx.y;
    const int slot = blockIdx.x;
    int jt;
    if (t == 0) {
        jt = slot;
    } else {
        int lst[4]; int cnt = 0;
        lst[cnt++] = 0;
        for (int kt = t - 1; kt <= t + 1; ++kt)
            if (kt >= 1 && kt <= 31) lst[cnt++] = kt;
        if (slot >= cnt) return;
        jt = lst[slot];
    }
    const int L = threadIdx.x & 31;
    const int wave = threadIdx.x >> 5;
    const int g = L >> 4, m = L & 15;

    for (int ii = 0; ii < 2; ++ii) {
        const int i = t * 16 + wave * 2 + ii;
        const int j = jt * 16 + m;  // A rows = 16 keys of this tile
        const float* prow = pair + ((size_t)i * NN + j) * CZ;
        CF c;
#pragma unroll
        for (int r = 0; r < 8; ++r) c.e[r] = 0.f;
#pragma unroll
        for (int cb = 0; cb < CZ; cb += 32) {
            AF a, b;
            f32x4 f0 = *(const f32x4*)(prow + cb + 8 * g);
            f32x4 f1 = *(const f32x4*)(prow + cb + 8 * g + 4);
            f32x4 f2 = *(const f32x4*)(prow + cb + 16 + 8 * g);
            f32x4 f3 = *(const f32x4*)(prow + cb + 16 + 8 * g + 4);
            a.u[0] = f2bf(f0.x); a.u[1] = f2bf(f0.y); a.u[2] = f2bf(f0.z); a.u[3] = f2bf(f0.w);
            a.u[4] = f2bf(f1.x); a.u[5] = f2bf(f1.y); a.u[6] = f2bf(f1.z); a.u[7] = f2bf(f1.w);
            a.u[8]  = f2bf(f2.x); a.u[9]  = f2bf(f2.y); a.u[10] = f2bf(f2.z); a.u[11] = f2bf(f2.w);
            a.u[12] = f2bf(f3.x); a.u[13] = f2bf(f3.y); a.u[14] = f2bf(f3.z); a.u[15] = f2bf(f3.w);
            b.q[0] = *(const u32x4*)(wbT + m * CZ + cb + 8 * g);
            b.q[1] = *(const u32x4*)(wbT + m * CZ + cb + 16 + 8 * g);
            c.v = __builtin_amdgcn_wmma_f32_16x16x32_bf16(false, a.v, false, b.v,
                                                          (short)0, c.v, false, false);
        }
        const int h = m;  // C cols = heads
        if (h < HH) {
#pragma unroll
            for (int r = 0; r < 8; ++r) {
                int jj = jt * 16 + r + 8 * g;
                bias_ws[((h * NN) + i) * NN + jj] = c.e[r] + b_b[h];
            }
        }
    }
}

// ---------------------------------------------------------------------------
// Kernel 4: sparse flash attention, wave == head. Key tiles processed in
// PAIRS so every PV WMMA runs full K=32 (odd leftover gets a fully-masked
// dummy partner -> p==0). All operand fragments are 2 x b128 loads.
// ---------------------------------------------------------------------------
__global__ void attn_kernel(const unsigned short* __restrict__ q_ws,
                            const unsigned short* __restrict__ k_ws,
                            const unsigned short* __restrict__ vT_ws,
                            const float* __restrict__ bias_ws,
                            unsigned short* attn_out) {
    __shared__ unsigned short p_lds[HH][16][32];
    const int t = blockIdx.x;
    const int s = blockIdx.y;
    const int h = threadIdx.x >> 5;  // wave = head
    const int L = threadIdx.x & 31;
    const int g = L >> 4, m = L & 15;

    auto pair_of = [&](int li, int& kt0, int& kt1, bool& val1) {
        if (t == 0)        { kt0 = 2 * li; kt1 = 2 * li + 1; val1 = true; }
        else if (li == 0)  { kt0 = 0; kt1 = (t == 1) ? 1 : (t - 1); val1 = true; }
        else {
            if (t == 1)       { kt0 = 2;  kt1 = 2;  val1 = false; }
            else if (t == 31) { kt0 = 31; kt1 = 31; val1 = false; }
            else              { kt0 = t;  kt1 = t + 1; val1 = true; }
        }
    };
    const int npairs = (t == 0) ? 16 : 2;

    // Q fragment (A-layout), pre-scaled by 1/sqrt(D)
    AF aq;
    const unsigned short* qbase = q_ws + ((s * HH + h) * NN + t * 16) * DD;
    aq.q[0] = *(const u32x4*)(qbase + m * DD + 16 * g);
    aq.q[1] = *(const u32x4*)(qbase + m * DD + 16 * g + 8);

    CF o0, o1;
    float mrow[8], lrow[8];
#pragma unroll
    for (int r = 0; r < 8; ++r) {
        o0.e[r] = 0.f; o1.e[r] = 0.f; mrow[r] = -3.0e38f; lrow[r] = 0.f;
    }

    const unsigned short* kbase = k_ws + (size_t)(s * HH + h) * NN * DD;
    const unsigned short* vT    = vT_ws + (size_t)(s * HH + h) * DD * NN;

    for (int li = 0; li < npairs; ++li) {
        int kt0, kt1; bool val1;
        pair_of(li, kt0, kt1, val1);
        if (li + 1 < npairs) {  // prefetch next pair's K tiles
            int nk0, nk1; bool nv;
            pair_of(li + 1, nk0, nk1, nv);
            __builtin_prefetch(kbase + (nk0 * 16 + m) * DD, 0, 0);
            __builtin_prefetch(kbase + (nk1 * 16 + m) * DD, 0, 0);
        }

        // score B fragments: B[K=d][N=key], contiguous along d
        AF bk0, bk1;
        {
            const unsigned short* kp0 = kbase + (kt0 * 16 + m) * DD + 16 * g;
            const unsigned short* kp1 = kbase + (kt1 * 16 + m) * DD + 16 * g;
            bk0.q[0] = *(const u32x4*)(kp0);
            bk0.q[1] = *(const u32x4*)(kp0 + 8);
            bk1.q[0] = *(const u32x4*)(kp1);
            bk1.q[1] = *(const u32x4*)(kp1 + 8);
        }
        CF sc0, sc1;
#pragma unroll
        for (int r = 0; r < 8; ++r) { sc0.e[r] = 0.f; sc1.e[r] = 0.f; }
        sc0.v = __builtin_amdgcn_wmma_f32_16x16x32_bf16(false, aq.v, false, bk0.v,
                                                        (short)0, sc0.v, false, false);
        sc1.v = __builtin_amdgcn_wmma_f32_16x16x32_bf16(false, aq.v, false, bk1.v,
                                                        (short)0, sc1.v, false, false);

        // bias + sparsity mask + online softmax over the 32-key pair
#pragma unroll
        for (int r = 0; r < 8; ++r) {
            const int i = t * 16 + r + 8 * g;
            const int j0 = kt0 * 16 + m;
            const int j1 = kt1 * 16 + m;
            float v0 = sc0.e[r] + bias_ws[((h * NN) + i) * NN + j0];
            float v1 = sc1.e[r] + bias_ws[((h * NN) + i) * NN + j1];
            int d0 = (i > j0) ? (i - j0) : (j0 - i);
            int d1 = (i > j1) ? (i - j1) : (j1 - i);
            bool a0 = (d0 <= 16) || (i < 16) || (j0 < 16);
            bool a1 = ((d1 <= 16) || (i < 16) || (j1 < 16)) && val1;
            v0 = a0 ? v0 : NEGV;
            v1 = a1 ? v1 : NEGV;

            float vmax = fmaxf(v0, v1);
            for (int msk = 1; msk < 16; msk <<= 1)
                vmax = fmaxf(vmax, __shfl_xor(vmax, msk, 32));
            float mnew = fmaxf(mrow[r], vmax);
            float alpha = __expf(mrow[r] - mnew);
            float p0 = __expf(v0 - mnew);
            float p1 = __expf(v1 - mnew);
            float rs = p0 + p1;
            for (int msk = 1; msk < 16; msk <<= 1)
                rs += __shfl_xor(rs, msk, 32);
            lrow[r] = lrow[r] * alpha + rs;
            o0.e[r] *= alpha; o1.e[r] *= alpha;
            mrow[r] = mnew;
            p_lds[h][r + 8 * g][m]      = f2bf(p0);
            p_lds[h][r + 8 * g][16 + m] = f2bf(p1);
        }
        asm volatile("s_wait_dscnt 0" ::: "memory");

        // P as A fragment (K=32, no padding), straight from LDS as b128s
        AF ap;
        ap.q[0] = *(const u32x4*)&p_lds[h][m][16 * g];
        ap.q[1] = *(const u32x4*)&p_lds[h][m][16 * g + 8];

        // V fragments from transposed V: contiguous along keys
        const int ktv = g ? kt1 : kt0;
        const unsigned short* vp0 = vT + (size_t)m * NN + ktv * 16;         // d = m
        const unsigned short* vp1 = vT + (size_t)(16 + m) * NN + ktv * 16;  // d = 16+m
        AF bv0, bv1;
        bv0.q[0] = *(const u32x4*)(vp0);
        bv0.q[1] = *(const u32x4*)(vp0 + 8);
        bv1.q[0] = *(const u32x4*)(vp1);
        bv1.q[1] = *(const u32x4*)(vp1 + 8);

        o0.v = __builtin_amdgcn_wmma_f32_16x16x32_bf16(false, ap.v, false, bv0.v,
                                                       (short)0, o0.v, false, false);
        o1.v = __builtin_amdgcn_wmma_f32_16x16x32_bf16(false, ap.v, false, bv1.v,
                                                       (short)0, o1.v, false, false);
    }

    // normalize + store bf16 attention output [s][n][h*32+d]
#pragma unroll
    for (int r = 0; r < 8; ++r) {
        int i = t * 16 + r + 8 * g;
        float inv = 1.0f / lrow[r];
        attn_out[((s * NN + i) * CM) + h * DD + m]      = f2bf(o0.e[r] * inv);
        attn_out[((s * NN + i) * CM) + h * DD + 16 + m] = f2bf(o1.e[r] * inv);
    }
}

// ---------------------------------------------------------------------------
// Kernel 5: out-projection WMMA (unrolled K=256) + residual + LayerNorm.
// ---------------------------------------------------------------------------
__global__ void outproj_ln_kernel(const unsigned short* __restrict__ attn_out,
                                  const unsigned short* __restrict__ woT,
                                  const float* __restrict__ b_o,
                                  const float* __restrict__ msa,
                                  const float* __restrict__ ln_g,
                                  const float* __restrict__ ln_b,
                                  float* __restrict__ out) {
    __shared__ float smem[16][CM];
    __shared__ float pS[16][16];
    __shared__ float pQ[16][16];
    __shared__ float muS[16], rstdS[16];

    const int rt = blockIdx.x;  // 0..1023
    const int wave = threadIdx.x >> 5;
    const int L = threadIdx.x & 31;
    const int g = L >> 4, m = L & 15;

    for (int cti = 0; cti < 2; ++cti) {
        const int ct = wave * 2 + cti;
        const int row = rt * 16 + m;
        const int col = ct * 16 + m;
        const unsigned short* arow = attn_out + row * CM;
        const unsigned short* brow = woT + col * CM;
        CF c;
#pragma unroll
        for (int r = 0; r < 8; ++r) c.e[r] = 0.f;
#pragma unroll
        for (int kb = 0; kb < CM; kb += 32) {
            AF a, b;
            a.q[0] = *(const u32x4*)(arow + kb + 8 * g);
            a.q[1] = *(const u32x4*)(arow + kb + 16 + 8 * g);
            b.q[0] = *(const u32x4*)(brow + kb + 8 * g);
            b.q[1] = *(const u32x4*)(brow + kb + 16 + 8 * g);
            c.v = __builtin_amdgcn_wmma_f32_16x16x32_bf16(false, a.v, false, b.v,
                                                          (short)0, c.v, false, false);
        }
#pragma unroll
        for (int r = 0; r < 8; ++r) smem[r + 8 * g][col] = c.e[r] + b_o[col];
    }
    __syncthreads();

    const int row_l = threadIdx.x >> 4;  // 0..15
    const int l16 = threadIdx.x & 15;
    const int grow = rt * 16 + row_l;
    float s1 = 0.f, s2 = 0.f;
    for (int cc = l16; cc < CM; cc += 16) {
        float y = smem[row_l][cc] + msa[grow * CM + cc];
        s1 += y; s2 += y * y;
    }
    pS[row_l][l16] = s1; pQ[row_l][l16] = s2;
    __syncthreads();
    if (l16 == 0) {
        float a = 0.f, b = 0.f;
#pragma unroll
        for (int k = 0; k < 16; ++k) { a += pS[row_l][k]; b += pQ[row_l][k]; }
        float mean = a * (1.f / CM);
        float var = b * (1.f / CM) - mean * mean;
        muS[row_l] = mean; rstdS[row_l] = rsqrtf(var + 1e-5f);
    }
    __syncthreads();
    const float mean = muS[row_l], rs = rstdS[row_l];
    for (int cc = l16; cc < CM; cc += 16) {
        float y = smem[row_l][cc] + msa[grow * CM + cc];
        out[grow * CM + cc] = (y - mean) * rs * ln_g[cc] + ln_b[cc];
    }
}

// ---------------------------------------------------------------------------
extern "C" void kernel_launch(void* const* d_in, const int* in_sizes, int n_in,
                              void* d_out, int out_size, void* d_ws, size_t ws_size,
                              hipStream_t stream) {
    (void)in_sizes; (void)n_in; (void)out_size; (void)ws_size;
    const float* msa  = (const float*)d_in[0];
    const float* pair = (const float*)d_in[1];
    const float* w_q  = (const float*)d_in[2];
    const float* b_q  = (const float*)d_in[3];
    const float* w_k  = (const float*)d_in[4];
    const float* b_k  = (const float*)d_in[5];
    const float* w_v  = (const float*)d_in[6];
    const float* b_v  = (const float*)d_in[7];
    const float* w_o  = (const float*)d_in[8];
    const float* b_o  = (const float*)d_in[9];
    const float* w_b  = (const float*)d_in[10];
    const float* b_b  = (const float*)d_in[11];
    const float* ln_g = (const float*)d_in[12];
    const float* ln_b = (const float*)d_in[13];
    float* out = (float*)d_out;

    char* ws = (char*)d_ws;
    size_t off = 0;
    auto alloc = [&](size_t bytes) -> void* {
        void* p = ws + off;
        off = (off + bytes + 255) & ~(size_t)255;
        return p;
    };
    unsigned short* msa_bf  = (unsigned short*)alloc((size_t)SS * NN * CM * 2);
    unsigned short* wqkvT   = (unsigned short*)alloc((size_t)3 * CM * CM * 2);
    unsigned short* woT     = (unsigned short*)alloc((size_t)CM * CM * 2);
    unsigned short* wbT     = (unsigned short*)alloc((size_t)16 * CZ * 2);
    unsigned short* q_ws    = (unsigned short*)alloc((size_t)SS * HH * NN * DD * 2);
    unsigned short* k_ws    = (unsigned short*)alloc((size_t)SS * HH * NN * DD * 2);
    unsigned short* vT_ws   = (unsigned short*)alloc((size_t)SS * HH * DD * NN * 2);
    float*          bias_ws = (float*)alloc((size_t)HH * NN * NN * 4);
    unsigned short* attn_ws = (unsigned short*)alloc((size_t)SS * NN * CM * 2);

    prep_kernel<<<dim3(2048), dim3(256), 0, stream>>>(
        msa, w_q, w_k, w_v, w_o, w_b, msa_bf, wqkvT, woT, wbT);

    qkv_gemm_kernel<<<dim3(1024, 6), dim3(256), 0, stream>>>(
        msa_bf, wqkvT, b_q, b_k, b_v, q_ws, k_ws, vT_ws);

    bias_kernel<<<dim3(32, 32), dim3(256), 0, stream>>>(
        pair, wbT, b_b, bias_ws);

    attn_kernel<<<dim3(32, 32), dim3(256), 0, stream>>>(
        q_ws, k_ws, vT_ws, bias_ws, attn_ws);

    outproj_ln_kernel<<<dim3(1024), dim3(256), 0, stream>>>(
        attn_ws, woT, b_o, msa, ln_g, ln_b, out);
}
